// RelationAwareAttention_1769526526264
// MI455X (gfx1250) — compile-verified, matrix operand order checked by reading, not run on hardware
//
#include <hip/hip_runtime.h>
#include <hip/hip_bf16.h>

// ---------------------------------------------------------------------------
// Relation-aware attention for MI455X (gfx1250, wave32, WMMA).
//
// Roofline: relation_weights (256 MB) dominates traffic -> memory bound at
// ~13us on 23.3 TB/s HBM *if* we never spill the 2x16x2048x2048 score tensor.
// Hence a fused flash-style attention kernel with online normalization, f16
// WMMA (16x16x32, K == head_dim == 32) for all GEMM work, f32 accumulate.
// ---------------------------------------------------------------------------

typedef __attribute__((ext_vector_type(16))) _Float16 v16h;
typedef __attribute__((ext_vector_type(8)))  _Float16 v8h;
typedef __attribute__((ext_vector_type(8)))  float    v8f;

#define E_DIM   512
#define NHEAD   16
#define RHEAD   8
#define NORMAL_HEADS 8
#define HEAD_DIM 32
#define BATCH   2
#define SEQ     2048
#define SCALING 0.17677669529663687f   // 32^-0.5

union FragU { v16h v; v8h h[2]; };

// A/B fragment loader (16-bit, 16x16x32 WMMA).
// A layout (ISA 7.12.2): lane = 16*hk + m holds row m; element e holds
//   k = (e<8 ? 0 : 16) + hk*8 + (e&7)  -> two contiguous 8-half (b128) loads.
// B fragments use the mirrored layout (lane&15 = column), so a row-major
// [N x K] (i.e. pre-transposed) matrix loads with the identical pattern.
__device__ __forceinline__ v16h load_frag(const _Float16* __restrict__ base,
                                          int ld, int row0, int k0, int lane) {
  const int m  = lane & 15;
  const int hk = lane >> 4;
  const _Float16* p = base + (size_t)(row0 + m) * ld + (k0 + hk * 8);
  FragU f;
  f.h[0] = *reinterpret_cast<const v8h*>(p);
  f.h[1] = *reinterpret_cast<const v8h*>(p + 16);
  return f.v;
}

__device__ __forceinline__ v8f wmma_f16(v16h a, v16h b, v8f c) {
  return __builtin_amdgcn_wmma_f32_16x16x32_f16(false, a, false, b,
                                                (short)0, c, false, false);
}

// ---------------------------------------------------------------------------
// Format converters (one pass; tiny vs. 256MB relation stream).
// ---------------------------------------------------------------------------
__global__ void cvt_f32_to_f16(const float* __restrict__ src,
                               _Float16* __restrict__ dst, int n) {
  int i = blockIdx.x * blockDim.x + threadIdx.x;
  if (i < n) dst[i] = (_Float16)src[i];
}

// WT[n][k] = W[k][n]  (so GEMM B-fragments are contiguous b128 loads)
__global__ void transpose_w_f16(const float* __restrict__ W,
                                _Float16* __restrict__ WT) {
  int i = blockIdx.x * blockDim.x + threadIdx.x;   // E*E threads
  int k = i >> 9, n = i & (E_DIM - 1);
  WT[(size_t)n * E_DIM + k] = (_Float16)W[i];
}

// ---------------------------------------------------------------------------
// Projection GEMM: Y = X @ W + b.  One 16x16 tile per wave, 8 waves/block.
// Output scattered to head-major f16: [B,H,L,HD] (q,k) or [B,H,HD,L] (v^T).
// ---------------------------------------------------------------------------
__global__ void __launch_bounds__(256)
proj_gemm(const _Float16* __restrict__ X16, const _Float16* __restrict__ WT16,
          const float* __restrict__ bias, _Float16* __restrict__ out_std,
          _Float16* __restrict__ out_vT, int write_vT) {
  const int lane = threadIdx.x & 31, wid = threadIdx.x >> 5;
  const int m = lane & 15, hk = lane >> 4;
  const int rbase = blockIdx.x * 16;
  const int cbase = blockIdx.y * 128 + wid * 16;
  v8f acc = {};
#pragma unroll
  for (int kk = 0; kk < E_DIM; kk += 32) {
    v16h a = load_frag(X16, E_DIM, rbase, kk, lane);
    v16h b = load_frag(WT16, E_DIM, cbase, kk, lane);
    acc = wmma_f16(a, b, acc);
  }
  const int col = cbase + m;
  const float bv = bias[col];
  const int h = col >> 5, d = col & (HEAD_DIM - 1);
#pragma unroll
  for (int r = 0; r < 8; ++r) {
    int row = rbase + hk * 8 + r;
    int bb = row >> 11, l = row & (SEQ - 1);
    float v = acc[r] + bv;
    if (write_vT) {
      out_vT[(((size_t)(bb * NHEAD + h)) * HEAD_DIM + d) * SEQ + l] = (_Float16)v;
    } else {
      out_std[(((size_t)(bb * NHEAD + h)) * SEQ + l) * HEAD_DIM + d] = (_Float16)v;
    }
  }
}

// ---------------------------------------------------------------------------
// Fused flash attention.  Wave owns 16 q-rows; block = 8 waves = 128 q-rows
// for one (b,h).  64 chunks of 32 keys: 2 WMMAs for scores, online max/sum
// (relation heads fold the streamed relation weights into p before the sum),
// P staged through LDS to convert C-layout -> A-layout, 2 WMMAs for P.V.
// ---------------------------------------------------------------------------
__global__ void __launch_bounds__(256)
attn_fused(const _Float16* __restrict__ qh, const _Float16* __restrict__ kh,
           const _Float16* __restrict__ vT, const float* __restrict__ relw,
           _Float16* __restrict__ attn_out) {
  const int lane = threadIdx.x & 31, wid = threadIdx.x >> 5;
  const int m = lane & 15, hk = lane >> 4;
  const int b = blockIdx.z, h = blockIdx.y;
  const int qbase = blockIdx.x * 128 + wid * 16;
  const size_t headoff = (size_t)(b * NHEAD + h);
  const _Float16* Q = qh + headoff * SEQ * HEAD_DIM;
  const _Float16* K = kh + headoff * SEQ * HEAD_DIM;
  const _Float16* V = vT + headoff * HEAD_DIM * SEQ;
  const bool is_rel = (h >= NORMAL_HEADS);
  const float* relbase =
      is_rel ? relw + ((size_t)(b * RHEAD + (h - NORMAL_HEADS))) * SEQ * SEQ
             : nullptr;

  __shared__ __align__(16) _Float16 ldsP[8][16][32];

  v16h aq = load_frag(Q, HEAD_DIM, qbase, 0, lane);
  v8f o0 = {}, o1 = {};
  float mrun[8], lrun[8];
#pragma unroll
  for (int r = 0; r < 8; ++r) { mrun[r] = -1e30f; lrun[r] = 0.0f; }

  for (int j = 0; j < SEQ; j += 32) {
    // scores: S[16 rows][32 keys] as two 16x16 C-layout tiles
    v16h bk0 = load_frag(K, HEAD_DIM, j, 0, lane);
    v16h bk1 = load_frag(K, HEAD_DIM, j + 16, 0, lane);
    v8f s0 = {}, s1 = {};
    s0 = wmma_f16(aq, bk0, s0);
    s1 = wmma_f16(aq, bk1, s1);

    if (is_rel && (j + 64) < SEQ) {  // stream-ahead the relation weights
      __builtin_prefetch(relbase + (size_t)(qbase + hk * 8) * SEQ + j + 64 + m, 0, 1);
    }

#pragma unroll
    for (int r = 0; r < 8; ++r) {
      float x0 = s0[r] * SCALING;
      float x1 = s1[r] * SCALING;
      float mx = fmaxf(x0, x1);
      mx = fmaxf(mx, __shfl_xor(mx, 1));
      mx = fmaxf(mx, __shfl_xor(mx, 2));
      mx = fmaxf(mx, __shfl_xor(mx, 4));
      mx = fmaxf(mx, __shfl_xor(mx, 8));        // row-max over 16 lanes
      float mnew  = fmaxf(mrun[r], mx);
      float alpha = __expf(mrun[r] - mnew);
      float p0 = __expf(x0 - mnew);
      float p1 = __expf(x1 - mnew);
      if (is_rel) {                             // exp(s)*w, renormalized online
        const float* rp = relbase + (size_t)(qbase + hk * 8 + r) * SEQ + j + m;
        p0 *= rp[0];
        p1 *= rp[16];
      }
      float rs = p0 + p1;
      rs += __shfl_xor(rs, 1);
      rs += __shfl_xor(rs, 2);
      rs += __shfl_xor(rs, 4);
      rs += __shfl_xor(rs, 8);
      lrun[r] = lrun[r] * alpha + rs;
      mrun[r] = mnew;
      o0[r] *= alpha;
      o1[r] *= alpha;
      ldsP[wid][hk * 8 + r][m]      = (_Float16)p0;   // C-layout -> LDS
      ldsP[wid][hk * 8 + r][16 + m] = (_Float16)p1;
    }
    __syncthreads();
    // read P back in A-layout (two ds b128 loads per lane)
    FragU pf;
    pf.h[0] = *reinterpret_cast<const v8h*>(&ldsP[wid][m][hk * 8]);
    pf.h[1] = *reinterpret_cast<const v8h*>(&ldsP[wid][m][16 + hk * 8]);
    // V^T rows are head dims -> contiguous-in-key b128 B-fragment loads
    v16h bv0 = load_frag(V, SEQ, 0, j, lane);
    v16h bv1 = load_frag(V, SEQ, 16, j, lane);
    o0 = wmma_f16(pf.v, bv0, o0);
    o1 = wmma_f16(pf.v, bv1, o1);
    __syncthreads();
  }

  const float eps = is_rel ? 1e-6f : 0.0f;
#pragma unroll
  for (int r = 0; r < 8; ++r) {
    int row = qbase + hk * 8 + r;
    float inv = 1.0f / (lrun[r] + eps);
    size_t base = ((size_t)(b * SEQ + row)) * E_DIM + h * HEAD_DIM;
    attn_out[base + m]      = (_Float16)(o0[r] * inv);
    attn_out[base + 16 + m] = (_Float16)(o1[r] * inv);
  }
}

// ---------------------------------------------------------------------------
// Output projection + LayerNorm fused: block computes a full 16x512 strip
// (8 waves x 4 column tiles), LN reduces each row from LDS.
// ---------------------------------------------------------------------------
__global__ void __launch_bounds__(256)
outproj_ln(const _Float16* __restrict__ attn16, const _Float16* __restrict__ WoT,
           const float* __restrict__ bo, const float* __restrict__ gamma,
           const float* __restrict__ beta, float* __restrict__ out) {
  const int lane = threadIdx.x & 31, wid = threadIdx.x >> 5;
  const int m = lane & 15, hk = lane >> 4;
  const int rbase = blockIdx.x * 16;
  __shared__ float rowbuf[16][E_DIM];
  v8f acc[4] = {{}, {}, {}, {}};
  for (int kk = 0; kk < E_DIM; kk += 32) {
    v16h a = load_frag(attn16, E_DIM, rbase, kk, lane);
#pragma unroll
    for (int t = 0; t < 4; ++t) {
      v16h bw = load_frag(WoT, E_DIM, wid * 64 + t * 16, kk, lane);
      acc[t] = wmma_f16(a, bw, acc[t]);
    }
  }
#pragma unroll
  for (int t = 0; t < 4; ++t) {
    int col = wid * 64 + t * 16 + m;
    float bv = bo[col];
#pragma unroll
    for (int r = 0; r < 8; ++r) rowbuf[hk * 8 + r][col] = acc[t][r] + bv;
  }
  __syncthreads();
#pragma unroll
  for (int rr = 0; rr < 2; ++rr) {
    int row = wid * 2 + rr;
    float s = 0.f, s2 = 0.f;
    for (int c = lane; c < E_DIM; c += 32) {
      float v = rowbuf[row][c];
      s += v; s2 += v * v;
    }
#pragma unroll
    for (int off = 16; off > 0; off >>= 1) {
      s  += __shfl_xor(s, off);
      s2 += __shfl_xor(s2, off);
    }
    float mean = s * (1.0f / E_DIM);
    float var  = s2 * (1.0f / E_DIM) - mean * mean;
    float rstd = rsqrtf(var + 1e-5f);
    for (int c = lane; c < E_DIM; c += 32) {
      out[(size_t)(rbase + row) * E_DIM + c] =
          (rowbuf[row][c] - mean) * rstd * gamma[c] + beta[c];
    }
  }
}

// ---------------------------------------------------------------------------
extern "C" void kernel_launch(void* const* d_in, const int* in_sizes, int n_in,
                              void* d_out, int out_size, void* d_ws, size_t ws_size,
                              hipStream_t stream) {
  const float* query = (const float*)d_in[0];
  const float* key   = (const float*)d_in[1];
  const float* value = (const float*)d_in[2];
  const float* relw  = (const float*)d_in[3];
  const float* Wq = (const float*)d_in[4];
  const float* bq = (const float*)d_in[5];
  const float* Wk = (const float*)d_in[6];
  const float* bk = (const float*)d_in[7];
  const float* Wv = (const float*)d_in[8];
  const float* bv = (const float*)d_in[9];
  const float* Wo = (const float*)d_in[10];
  const float* bo = (const float*)d_in[11];
  const float* gamma = (const float*)d_in[12];
  const float* beta  = (const float*)d_in[13];
  float* out = (float*)d_out;

  // workspace carve-up (~30 MB)
  const size_t XE = (size_t)BATCH * SEQ * E_DIM;   // 2M elems
  const size_t WE = (size_t)E_DIM * E_DIM;         // 256K elems
  char* ws = (char*)d_ws;
  _Float16* xq16 = (_Float16*)ws; ws += XE * sizeof(_Float16);
  _Float16* xk16 = (_Float16*)ws; ws += XE * sizeof(_Float16);
  _Float16* xv16 = (_Float16*)ws; ws += XE * sizeof(_Float16);
  _Float16* WqT  = (_Float16*)ws; ws += WE * sizeof(_Float16);
  _Float16* WkT  = (_Float16*)ws; ws += WE * sizeof(_Float16);
  _Float16* WvT  = (_Float16*)ws; ws += WE * sizeof(_Float16);
  _Float16* WoT  = (_Float16*)ws; ws += WE * sizeof(_Float16);
  _Float16* qh16 = (_Float16*)ws; ws += XE * sizeof(_Float16);
  _Float16* kh16 = (_Float16*)ws; ws += XE * sizeof(_Float16);
  _Float16* vT16 = (_Float16*)ws; ws += XE * sizeof(_Float16);
  _Float16* at16 = (_Float16*)ws; ws += XE * sizeof(_Float16);
  (void)in_sizes; (void)n_in; (void)out_size; (void)ws_size;

  dim3 blk(256);
  const int nX = (int)XE;
  cvt_f32_to_f16<<<(nX + 255) / 256, blk, 0, stream>>>(query, xq16, nX);
  cvt_f32_to_f16<<<(nX + 255) / 256, blk, 0, stream>>>(key,   xk16, nX);
  cvt_f32_to_f16<<<(nX + 255) / 256, blk, 0, stream>>>(value, xv16, nX);
  const int nW = (int)WE;
  transpose_w_f16<<<nW / 256, blk, 0, stream>>>(Wq, WqT);
  transpose_w_f16<<<nW / 256, blk, 0, stream>>>(Wk, WkT);
  transpose_w_f16<<<nW / 256, blk, 0, stream>>>(Wv, WvT);
  transpose_w_f16<<<nW / 256, blk, 0, stream>>>(Wo, WoT);

  dim3 pg(BATCH * SEQ / 16, E_DIM / 128);            // (256, 4)
  proj_gemm<<<pg, blk, 0, stream>>>(xq16, WqT, bq, qh16, nullptr, 0);
  proj_gemm<<<pg, blk, 0, stream>>>(xk16, WkT, bk, kh16, nullptr, 0);
  proj_gemm<<<pg, blk, 0, stream>>>(xv16, WvT, bv, nullptr, vT16, 1);

  dim3 ag(SEQ / 128, NHEAD, BATCH);                  // (16, 16, 2)
  attn_fused<<<ag, blk, 0, stream>>>(qh16, kh16, vT16, relw, at16);

  outproj_ln<<<BATCH * SEQ / 16, blk, 0, stream>>>(at16, WoT, bo, gamma, beta, out);
}
